// rAttention_57784490001264
// MI455X (gfx1250) — matrix operand order repaired and checked by reading, not compiled, not run on hardware
//
#include <hip/hip_runtime.h>
#include <hip/hip_bf16.h>
#include <math.h>

typedef __attribute__((ext_vector_type(8)))  float   v8f;
typedef __attribute__((ext_vector_type(16))) __bf16  v16bf;
typedef __attribute__((ext_vector_type(8)))  __bf16  bf8v;
typedef __attribute__((ext_vector_type(4)))  __bf16  bf4v;

#define W_WIN   21
#define W_HALF  10
#define N_HEADS 8
#define HEAD_DIM 64

// ---------------------------------------------------------------------------
// Split-precision GEMM: C[M,N] = A[M,K]*B[K,N] + bias[N]  (near-fp32 accuracy)
// A,B decomposed to bf16 hi+lo during tile load; 3 x v_wmma_f32_16x16x32_bf16
// per K=32 step: hi*hi + hi*lo + lo*hi (fp32 accumulate).
// Block = 256 threads (8 wave32). Tile = 128(M) x 64(N), K chunk = 32.
// Each wave owns a 16(M) x 64(N) strip = 4 v8f accumulators.
// ---------------------------------------------------------------------------
#define TM 128
#define TN 64
#define TK 32
#define ASTR 40   // bf16 elements per LDS row (80B: 16B-aligned, conflict-free)

__device__ inline v16bf ld_frag16(const __bf16* p) {
    bf8v lo = *(const bf8v*)(p);
    bf8v hi = *(const bf8v*)(p + 8);
    return __builtin_shufflevector(lo, hi, 0,1,2,3,4,5,6,7,8,9,10,11,12,13,14,15);
}

__global__ __launch_bounds__(256) void wmma_gemm_kernel(
    const float* __restrict__ A, const float* __restrict__ B,
    const float* __restrict__ bias, float* __restrict__ C,
    int M, int N, int K)
{
    __shared__ __bf16 Ah[TM][ASTR];
    __shared__ __bf16 Al[TM][ASTR];
    __shared__ __bf16 Bth[TN][ASTR];   // transposed: [col][k]
    __shared__ __bf16 Btl[TN][ASTR];

    const int tid   = threadIdx.x;
    const int lane  = tid & 31;
    const int wave  = tid >> 5;              // 0..7
    const int mBase = blockIdx.y * TM;
    const int nBase = blockIdx.x * TN;
    const int half  = lane >> 4;             // 0: lanes 0-15, 1: lanes 16-31
    const int l16   = lane & 15;
    const int mrow  = wave * 16 + l16;

    v8f acc[4];
    #pragma unroll
    for (int t = 0; t < 4; ++t) acc[t] = (v8f)0.0f;

    const int bcol = tid & 63;               // B loader: column
    const int bkr  = (tid >> 6) * 8;         // B loader: k-range start

    for (int k0 = 0; k0 < K; k0 += TK) {
        // ---- A tile: 128x32 fp32 -> hi/lo bf16, K in natural order so a
        //      lane's 16 values (K h*16..h*16+15) are contiguous.
        #pragma unroll
        for (int i = 0; i < 4; ++i) {
            int idx = tid + i * 256;
            int row = idx >> 3;              // 8 float4 per 32-wide row
            int c4  = idx & 7;
            float4 v = *reinterpret_cast<const float4*>(
                A + (size_t)(mBase + row) * K + (k0 + c4 * 4));
            int c4p = c4;  // natural K order
            bf4v h, l;
            h[0] = (__bf16)v.x; l[0] = (__bf16)(v.x - (float)h[0]);
            h[1] = (__bf16)v.y; l[1] = (__bf16)(v.y - (float)h[1]);
            h[2] = (__bf16)v.z; l[2] = (__bf16)(v.z - (float)h[2]);
            h[3] = (__bf16)v.w; l[3] = (__bf16)(v.w - (float)h[3]);
            *reinterpret_cast<bf4v*>(&Ah[row][c4p * 4]) = h;
            *reinterpret_cast<bf4v*>(&Al[row][c4p * 4]) = l;
        }
        // ---- B tile: 32x64 fp32 -> transposed hi/lo bf16 Bt[col][k].
        //      Each thread: one column, 8 k's (coalesced across lanes).
        {
            float vv[8];
            #pragma unroll
            for (int j = 0; j < 8; ++j)
                vv[j] = B[(size_t)(k0 + bkr + j) * N + (nBase + bcol)];
            bf8v h, l;
            #pragma unroll
            for (int j = 0; j < 8; ++j) {
                __bf16 hb = (__bf16)vv[j];
                h[j] = hb;
                l[j] = (__bf16)(vv[j] - (float)hb);
            }
            *reinterpret_cast<bf8v*>(&Bth[bcol][bkr]) = h;
            *reinterpret_cast<bf8v*>(&Btl[bcol][bkr]) = l;
        }
        __syncthreads();

        // ---- compute: per n-tile, 3 bf16 WMMAs (hi*hi + hi*lo + lo*hi) ----
        v16bf ah = ld_frag16(&Ah[mrow][half * 16]);
        v16bf al = ld_frag16(&Al[mrow][half * 16]);
        #pragma unroll
        for (int t = 0; t < 4; ++t) {
            const int col = t * 16 + l16;
            v16bf bh = ld_frag16(&Bth[col][half * 16]);
            v16bf bl = ld_frag16(&Btl[col][half * 16]);
            acc[t] = __builtin_amdgcn_wmma_f32_16x16x32_bf16(
                false, ah, false, bh, (short)0, acc[t], false, false);
            acc[t] = __builtin_amdgcn_wmma_f32_16x16x32_bf16(
                false, ah, false, bl, (short)0, acc[t], false, false);
            acc[t] = __builtin_amdgcn_wmma_f32_16x16x32_bf16(
                false, al, false, bh, (short)0, acc[t], false, false);
        }
        __syncthreads();
    }

    // ---- store: C/D layout: vgpr r -> M=r (lanes 0-15) / M=r+8 (lanes 16-31)
    #pragma unroll
    for (int t = 0; t < 4; ++t) {
        const int col = nBase + t * 16 + l16;
        const float bv = bias[col];
        #pragma unroll
        for (int r = 0; r < 8; ++r) {
            const int row = mBase + wave * 16 + r + half * 8;
            C[(size_t)row * N + col] = acc[t][r] + bv;
        }
    }
}

// ---------------------------------------------------------------------------
// Banded attention: one wave32 per (batch, head, position).
// Each lane holds 2 of the 64 head dims. Scores via wave shuffle reduction.
// qkv layout: [B, L, 1536] rows = [q(512) | k(512) | v(512)].
// ---------------------------------------------------------------------------
__device__ inline float wave_sum32(float v) {
    #pragma unroll
    for (int m = 16; m >= 1; m >>= 1) v += __shfl_xor(v, m, 32);
    return v;
}

__global__ __launch_bounds__(256) void band_attn_kernel(
    const float* __restrict__ qkv, const float* __restrict__ bias_rw,
    float* __restrict__ z, int L)
{
    const int lane = threadIdx.x & 31;
    const int wid  = blockIdx.x * 8 + (threadIdx.x >> 5);

    const int p  = wid % L;
    const int bh = wid / L;
    const int h  = bh % N_HEADS;
    const int b  = bh / N_HEADS;

    const size_t rowStride = 3 * 512;
    const size_t base = ((size_t)b * L) * rowStride;
    const int dimOff = h * HEAD_DIM + lane * 2;

    const float2 q  = *reinterpret_cast<const float2*>(
        qkv + base + (size_t)p * rowStride + dimOff);
    const float2 br = *reinterpret_cast<const float2*>(
        bias_rw + h * HEAD_DIM + lane * 2);
    const float qx = q.x * 0.125f + br.x;   // 1/sqrt(64) = 0.125
    const float qy = q.y * 0.125f + br.y;

    const float NEG = -3.402823466e38f;
    float sc[W_WIN];

    #pragma unroll
    for (int o = 0; o < W_WIN; ++o) {
        const int pos = p + o - W_HALF;
        if (pos >= 0 && pos < L) {
            const float2 k = *reinterpret_cast<const float2*>(
                qkv + base + (size_t)pos * rowStride + 512 + dimOff);
            sc[o] = wave_sum32(qx * k.x + qy * k.y);
        } else {
            sc[o] = NEG;
        }
    }

    float mx = sc[0];
    #pragma unroll
    for (int o = 1; o < W_WIN; ++o) mx = fmaxf(mx, sc[o]);
    float den = 0.0f;
    #pragma unroll
    for (int o = 0; o < W_WIN; ++o) { sc[o] = __expf(sc[o] - mx); den += sc[o]; }
    const float inv = 1.0f / den;

    float zx = 0.0f, zy = 0.0f;
    #pragma unroll
    for (int o = 0; o < W_WIN; ++o) {
        const int pos = p + o - W_HALF;
        if (pos >= 0 && pos < L) {
            const float w = sc[o] * inv;
            const float2 v = *reinterpret_cast<const float2*>(
                qkv + base + (size_t)pos * rowStride + 1024 + dimOff);
            zx += w * v.x;
            zy += w * v.y;
        }
    }

    float2 out = make_float2(zx, zy);
    *reinterpret_cast<float2*>(z + ((size_t)b * L + p) * 512 + dimOff) = out;
}

// ---------------------------------------------------------------------------
// Launch: GEMM(qkv) -> banded attention -> GEMM(out)
// ---------------------------------------------------------------------------
extern "C" void kernel_launch(void* const* d_in, const int* in_sizes, int n_in,
                              void* d_out, int out_size, void* d_ws, size_t ws_size,
                              hipStream_t stream) {
    const float* x       = (const float*)d_in[0];
    const float* Wqkv    = (const float*)d_in[1];
    const float* bqkv    = (const float*)d_in[2];
    const float* bias_rw = (const float*)d_in[3];
    const float* Wout    = (const float*)d_in[4];
    const float* bout    = (const float*)d_in[5];
    float* out = (float*)d_out;

    const int Bn = 16;          // b*c = 8*2
    const int L  = 2048;
    const int D  = 512;
    const int M  = Bn * L;      // 32768

    float* qkv = (float*)d_ws;                       // [M, 1536]
    float* z   = qkv + (size_t)M * (3 * D);          // [M, 512]

    // 1) qkv = x @ Wqkv + bqkv
    {
        dim3 grid((3 * D) / TN, M / TM);
        wmma_gemm_kernel<<<grid, 256, 0, stream>>>(x, Wqkv, bqkv, qkv, M, 3 * D, D);
    }
    // 2) banded attention -> z
    {
        const int totalWaves = Bn * N_HEADS * L;     // 262144
        band_attn_kernel<<<totalWaves / 8, 256, 0, stream>>>(qkv, bias_rw, z, L);
    }
    // 3) out = z @ Wout + bout
    {
        dim3 grid(D / TN, M / TM);
        wmma_gemm_kernel<<<grid, 256, 0, stream>>>(z, Wout, bout, out, M, D, D);
    }
}
